// MFeature_23278722744353
// MI455X (gfx1250) — compile-verified, matrix operand order checked by reading, not run on hardware
//
#include <hip/hip_runtime.h>
#include <math.h>

// ---------------------------------------------------------------------------
// Types for CDNA5 WMMA (wave32): D(16x16 f32) = A(16x32 f16) * B(32x16 f16) + C
// ---------------------------------------------------------------------------
typedef __attribute__((ext_vector_type(16))) _Float16 v16h;
typedef __attribute__((ext_vector_type(8)))  float    v8f;
typedef __attribute__((ext_vector_type(4)))  _Float16 h4;

union AFrag { v16h v; _Float16 h[16]; };
union CFrag { v8f  v; float    f[8];  };
union H4    { h4   v; _Float16 h[4];  };

// Problem constants (match the reference)
static constexpr int cB   = 8;
static constexpr int cV   = 1024;
static constexpr int cD   = 512;
static constexpr int cDE  = 64;
static constexpr int cK   = 32;
static constexpr int cDFF = 2048;

__device__ __forceinline__ float gelu_erf(float x) {
    return 0.5f * x * (1.0f + erff(x * 0.70710678118654752f));
}

// ---------------------------------------------------------------------------
// WMMA GEMM:  C[M,N] = act(A[M,Kd] @ W[Kd,N] + bias) + res
// Requirements (hold for every call site): Kd % 32 == 0, N % 64 == 0.
// M may be arbitrary: A-tile rows are index-clamped (garbage rows feed only
// output rows that the guarded epilogue never stores).
// Tile 64x64, 128 threads (4 waves); all staging is vectorized & branch-free.
// ---------------------------------------------------------------------------
template <int ACT, bool RES> // ACT: 0 none, 1 exact GELU
__global__ void __launch_bounds__(128)
k_gemm(const float* __restrict__ A, const float* __restrict__ W,
       const float* __restrict__ bias, const float* __restrict__ res,
       float* __restrict__ C, int M, int Kd, int N)
{
    __shared__ _Float16 As[64][40];   // [m][k], 80B rows (16B aligned)
    __shared__ _Float16 Bs[64][40];   // [n][k]  (W tile stored transposed)

    const int tid  = threadIdx.x;
    const int wave = tid >> 5;
    const int lane = tid & 31;
    const int rlo  = lane & 15;
    const int half = lane >> 4;
    const int m0   = blockIdx.y * 64;
    const int n0   = blockIdx.x * 64;

    CFrag acc[4];
    for (int c = 0; c < 4; ++c)
        for (int r = 0; r < 8; ++r) acc[c].f[r] = 0.0f;

    for (int kk = 0; kk < Kd; kk += 32) {
        // A tile 64x32: float4 loads, rows clamped to [0, M-1]
        for (int it = 0; it < 4; ++it) {
            int idx = tid + it * 128;        // 0..511 float4s
            int r  = idx >> 3;               // 8 float4 per 32-wide row
            int cb = (idx & 7) * 4;
            int gm = m0 + r; gm = gm < M ? gm : (M - 1);
            float4 u = *(const float4*)(A + (size_t)gm * Kd + kk + cb);
            H4 hv;
            hv.h[0] = (_Float16)u.x; hv.h[1] = (_Float16)u.y;
            hv.h[2] = (_Float16)u.z; hv.h[3] = (_Float16)u.w;
            *(h4*)&As[r][cb] = hv.v;
        }
        // W tile 32x64: float4 loads (coalesced along N), transposed into LDS
        for (int it = 0; it < 4; ++it) {
            int idx = tid + it * 128;        // 0..511 float4s
            int kc = idx >> 4;               // 16 float4 per 64-wide row
            int nb = (idx & 15) * 4;
            float4 u = *(const float4*)(W + (size_t)(kk + kc) * N + n0 + nb);
            Bs[nb + 0][kc] = (_Float16)u.x;
            Bs[nb + 1][kc] = (_Float16)u.y;
            Bs[nb + 2][kc] = (_Float16)u.z;
            Bs[nb + 3][kc] = (_Float16)u.w;
        }
        __syncthreads();

        AFrag af;
        for (int i = 0; i < 16; ++i) {
            int g = i >> 3, w = i & 7;
            af.h[i] = As[wave * 16 + rlo][g * 16 + half * 8 + w];
        }
        AFrag bf[4];
        for (int c = 0; c < 4; ++c)
            for (int i = 0; i < 16; ++i) {
                int g = i >> 3, w = i & 7;
                bf[c].h[i] = Bs[c * 16 + rlo][g * 16 + half * 8 + w];
            }
        for (int c = 0; c < 4; ++c)
            acc[c].v = __builtin_amdgcn_wmma_f32_16x16x32_f16(
                false, af.v, false, bf[c].v, (short)0, acc[c].v, false, false);
        __syncthreads();
    }

    for (int c = 0; c < 4; ++c)
        for (int r = 0; r < 8; ++r) {
            int gm = m0 + wave * 16 + r + 8 * half;
            int gn = n0 + c * 16 + rlo;
            if (gm < M) {
                float v = acc[c].f[r];
                if (bias) v += bias[gn];
                if (ACT == 1) v = gelu_erf(v);
                if (RES) v += res[(size_t)gm * N + gn];
                C[(size_t)gm * N + gn] = v;
            }
        }
}

// ---------------------------------------------------------------------------
// Flash attention. 4 waves per block; each wave owns one 16-row query tile,
// all waves share the K/V tile staged in LDS (vectorized f32->f16 loads).
// Requires Nkv % 32 == 0 (true for all uses: 1024 and 32).
// MASKED: additive log-mask [Bn, Nq, Nkv] shared across heads (finite values).
// ---------------------------------------------------------------------------
template <int DK, bool MASKED>
__global__ void __launch_bounds__(128)
k_attn(const float* __restrict__ Q, const float* __restrict__ Km,
       const float* __restrict__ Vm, const float* __restrict__ mask,
       float* __restrict__ O, int Bn, int H, int Nq, int Nkv,
       float scale, float eps)
{
    constexpr int NQF = DK / 32;   // A-frags holding Q
    constexpr int NOC = DK / 16;   // output accumulators
    constexpr int KF4 = 32 * DK / 4;
    __shared__ _Float16 Ks[32][DK + 8];   // [j][d]   row = DK+8 halves (16B mult)
    __shared__ _Float16 Vs[DK][40];       // [od][j]  transposed, 80B rows
    __shared__ _Float16 Pl[4][16][40];    // per-wave P scratch, 80B rows

    const int tid  = threadIdx.x;
    const int wave = tid >> 5;
    const int lane = tid & 31;
    const int rlo  = lane & 15;
    const int half = lane >> 4;
    const int qt = blockIdx.x * 4 + wave;
    const int h = blockIdx.y, b = blockIdx.z;
    const int Dm = H * DK;

    const float* Qb = Q  + ((size_t)b * Nq)  * Dm + h * DK;
    const float* Kb = Km + ((size_t)b * Nkv) * Dm + h * DK;
    const float* Vb = Vm + ((size_t)b * Nkv) * Dm + h * DK;

    // Q fragments: aligned float4 pairs per 8-wide run (row clamped; garbage
    // rows are never stored)
    AFrag qf[NQF];
    {
        int qi = qt * 16 + rlo;
        const float* qrow = Qb + (size_t)(qi < Nq ? qi : (Nq - 1)) * Dm;
        for (int f = 0; f < NQF; ++f)
            for (int g = 0; g < 2; ++g) {
                const float4* p = (const float4*)(qrow + f * 32 + g * 16 + half * 8);
                float4 u0 = p[0], u1 = p[1];
                qf[f].h[g * 8 + 0] = (_Float16)(u0.x * scale);
                qf[f].h[g * 8 + 1] = (_Float16)(u0.y * scale);
                qf[f].h[g * 8 + 2] = (_Float16)(u0.z * scale);
                qf[f].h[g * 8 + 3] = (_Float16)(u0.w * scale);
                qf[f].h[g * 8 + 4] = (_Float16)(u1.x * scale);
                qf[f].h[g * 8 + 5] = (_Float16)(u1.y * scale);
                qf[f].h[g * 8 + 6] = (_Float16)(u1.z * scale);
                qf[f].h[g * 8 + 7] = (_Float16)(u1.w * scale);
            }
    }

    CFrag oacc[NOC];
    for (int c = 0; c < NOC; ++c)
        for (int r = 0; r < 8; ++r) oacc[c].f[r] = 0.0f;
    float mrow[8], lrow[8];
    for (int r = 0; r < 8; ++r) { mrow[r] = -1e30f; lrow[r] = 0.0f; }

    for (int j0 = 0; j0 < Nkv; j0 += 32) {
        // Cooperative tile staging (branch-free, f32->f16)
        for (int idx = tid; idx < KF4; idx += 128) {
            int j  = idx / (DK / 4);
            int cb = (idx % (DK / 4)) * 4;
            float4 u = *(const float4*)(Kb + (size_t)(j0 + j) * Dm + cb);
            H4 hv;
            hv.h[0] = (_Float16)u.x; hv.h[1] = (_Float16)u.y;
            hv.h[2] = (_Float16)u.z; hv.h[3] = (_Float16)u.w;
            *(h4*)&Ks[j][cb] = hv.v;
        }
        for (int idx = tid; idx < KF4; idx += 128) {
            int j  = idx / (DK / 4);
            int cb = (idx % (DK / 4)) * 4;
            float4 u = *(const float4*)(Vb + (size_t)(j0 + j) * Dm + cb);
            Vs[cb + 0][j] = (_Float16)u.x;
            Vs[cb + 1][j] = (_Float16)u.y;
            Vs[cb + 2][j] = (_Float16)u.z;
            Vs[cb + 3][j] = (_Float16)u.w;
        }
        __syncthreads();

        // S = Q K^T : prefetch all K fragments, then back-to-back WMMAs
        AFrag bk[2 * NQF];
        for (int f = 0; f < NQF; ++f)
            for (int i = 0; i < 16; ++i) {
                int g = i >> 3, w = i & 7;
                int kd = f * 32 + g * 16 + half * 8 + w;
                bk[2 * f + 0].h[i] = Ks[rlo][kd];
                bk[2 * f + 1].h[i] = Ks[16 + rlo][kd];
            }
        CFrag s0, s1;
        for (int r = 0; r < 8; ++r) { s0.f[r] = 0.0f; s1.f[r] = 0.0f; }
        for (int f = 0; f < NQF; ++f) {
            s0.v = __builtin_amdgcn_wmma_f32_16x16x32_f16(
                false, qf[f].v, false, bk[2 * f + 0].v, (short)0, s0.v, false, false);
            s1.v = __builtin_amdgcn_wmma_f32_16x16x32_f16(
                false, qf[f].v, false, bk[2 * f + 1].v, (short)0, s1.v, false, false);
        }

        if (MASKED) {
            const int c0 = j0 + rlo, c1 = j0 + 16 + rlo;
            for (int r = 0; r < 8; ++r) {
                int qi = qt * 16 + r + 8 * half;
                qi = qi < Nq ? qi : (Nq - 1);   // clamped; garbage rows unused
                const float* mp = mask + ((size_t)b * Nq + qi) * Nkv;
                s0.f[r] += mp[c0];
                s1.f[r] += mp[c1];
            }
        }

        float alpha[8];
        for (int r = 0; r < 8; ++r) {
            float mx = fmaxf(s0.f[r], s1.f[r]);
            for (int off = 1; off < 16; off <<= 1)
                mx = fmaxf(mx, __shfl_xor(mx, off, 32));
            float mn = fmaxf(mrow[r], mx);
            alpha[r] = expf(mrow[r] - mn);
            float p0 = expf(s0.f[r] - mn);
            float p1 = expf(s1.f[r] - mn);
            float ps = p0 + p1;
            for (int off = 1; off < 16; off <<= 1)
                ps += __shfl_xor(ps, off, 32);
            lrow[r] = lrow[r] * alpha[r] + ps;
            mrow[r] = mn;
            Pl[wave][r + 8 * half][rlo]      = (_Float16)p0;
            Pl[wave][r + 8 * half][16 + rlo] = (_Float16)p1;
        }
        // DS ops are in-order within a wave: safe to read own Pl slab now.
        AFrag pa;
        for (int i = 0; i < 16; ++i) {
            int g = i >> 3, w = i & 7;
            pa.h[i] = Pl[wave][rlo][g * 16 + half * 8 + w];
        }

        // O += P V : rescale, prefetch all V fragments, then WMMAs
        for (int c = 0; c < NOC; ++c)
            for (int r = 0; r < 8; ++r) oacc[c].f[r] *= alpha[r];
        AFrag vb[NOC];
        for (int c = 0; c < NOC; ++c)
            for (int i = 0; i < 16; ++i) {
                int g = i >> 3, w = i & 7;
                vb[c].h[i] = Vs[c * 16 + rlo][g * 16 + half * 8 + w];
            }
        for (int c = 0; c < NOC; ++c)
            oacc[c].v = __builtin_amdgcn_wmma_f32_16x16x32_f16(
                false, pa.v, false, vb[c].v, (short)0, oacc[c].v, false, false);
        __syncthreads();
    }

    float* Ob = O + ((size_t)b * Nq) * Dm + h * DK;
    for (int c = 0; c < NOC; ++c)
        for (int r = 0; r < 8; ++r) {
            int qi = qt * 16 + r + 8 * half;
            if (qi < Nq)
                Ob[(size_t)qi * Dm + c * 16 + rlo] = oacc[c].f[r] / (lrow[r] + eps);
        }
}

// ---------------------------------------------------------------------------
// log(clip(assign @ assign^T, 0, 1) + 1e-6)  -> logconn [B,V,V]
// ---------------------------------------------------------------------------
__global__ void k_conn(const float* __restrict__ assign, float* __restrict__ logconn,
                       int Bn, int Vn, int Kn)
{
    size_t idx = (size_t)blockIdx.x * blockDim.x + threadIdx.x;
    size_t total = (size_t)Bn * Vn * Vn;
    if (idx >= total) return;
    int w = (int)(idx % Vn);
    size_t t = idx / Vn;
    int v = (int)(t % Vn);
    int b = (int)(t / Vn);
    const float* av = assign + ((size_t)b * Vn + v) * Kn;
    const float* aw = assign + ((size_t)b * Vn + w) * Kn;
    float s = 0.0f;
    for (int k = 0; k < Kn; ++k) s += av[k] * aw[k];
    s = fminf(fmaxf(s, 0.0f), 1.0f);
    logconn[idx] = logf(s + 1e-6f);
}

// log(clip(assign[b,j,kq],0,1)+1e-6) -> lm [B,K,V]
__global__ void k_clsmask(const float* __restrict__ assign, float* __restrict__ lm,
                          int Bn, int Kn, int Vn)
{
    size_t idx = (size_t)blockIdx.x * blockDim.x + threadIdx.x;
    size_t total = (size_t)Bn * Kn * Vn;
    if (idx >= total) return;
    int j = (int)(idx % Vn);
    size_t t = idx / Vn;
    int kq = (int)(t % Kn);
    int b  = (int)(t / Kn);
    float a = assign[((size_t)b * Vn + j) * Kn + kq];
    a = fminf(fmaxf(a, 0.0f), 1.0f);
    lm[idx] = logf(a + 1e-6f);
}

// LayerNorm (+ optional FiLM: film row = [g1(D) | b1(D)]): one block per row.
__global__ void k_ln(const float* __restrict__ X, const float* __restrict__ g,
                     const float* __restrict__ b, const float* __restrict__ film,
                     float* __restrict__ Y, int M, int Dn)
{
    int row = blockIdx.x;
    if (row >= M) return;
    const float* x = X + (size_t)row * Dn;
    __shared__ float red[128];
    int tid = threadIdx.x;

    float s = 0.0f;
    for (int i = tid; i < Dn; i += 128) s += x[i];
    red[tid] = s; __syncthreads();
    for (int st = 64; st > 0; st >>= 1) { if (tid < st) red[tid] += red[tid + st]; __syncthreads(); }
    float mu = red[0] / (float)Dn;
    __syncthreads();

    float v = 0.0f;
    for (int i = tid; i < Dn; i += 128) { float d = x[i] - mu; v += d * d; }
    red[tid] = v; __syncthreads();
    for (int st = 64; st > 0; st >>= 1) { if (tid < st) red[tid] += red[tid + st]; __syncthreads(); }
    float rinv = rsqrtf(red[0] / (float)Dn + 1e-5f);

    const float* f = film ? (film + (size_t)row * 2 * Dn) : nullptr;
    for (int i = tid; i < Dn; i += 128) {
        float y = (x[i] - mu) * rinv * g[i] + b[i];
        if (f) y = y * (1.0f + f[i]) + f[Dn + i];
        Y[(size_t)row * Dn + i] = y;
    }
}

__global__ void k_bcast(float* __restrict__ out, const float* __restrict__ in,
                        size_t total, int inner)
{
    size_t i = (size_t)blockIdx.x * blockDim.x + threadIdx.x;
    if (i < total) out[i] = in[i % (size_t)inner];
}

// m[b,kq,:] = clsB[b,kq,:] where count(b,kq) <= 1e-6
__global__ void k_empty_fix(const float* __restrict__ assign, const float* __restrict__ clsB,
                            float* __restrict__ m, int Bn, int Kn, int Vn, int Dn)
{
    int idx = blockIdx.x;
    int b = idx / Kn, kq = idx % Kn;
    __shared__ float red[128];
    int tid = threadIdx.x;
    float s = 0.0f;
    for (int j = tid; j < Vn; j += 128) s += assign[((size_t)b * Vn + j) * Kn + kq];
    red[tid] = s; __syncthreads();
    for (int st = 64; st > 0; st >>= 1) { if (tid < st) red[tid] += red[tid + st]; __syncthreads(); }
    if (red[0] <= 1e-6f) {
        for (int i = tid; i < Dn; i += 128) {
            size_t o = ((size_t)b * Kn + kq) * Dn + i;
            m[o] = clsB[o];
        }
    }
}

// ---------------------------------------------------------------------------
// Host side
// ---------------------------------------------------------------------------
struct LinP { const float* b; const float* w; };
struct LNPm { const float* b; const float* g; };

static void gemm(hipStream_t st, const float* A, const float* w, const float* b,
                 const float* res, float* C, int M, int Kd, int N, int act)
{
    dim3 g((N + 63) / 64, (M + 63) / 64);
    if (act)      k_gemm<1, false><<<g, dim3(128), 0, st>>>(A, w, b, nullptr, C, M, Kd, N);
    else if (res) k_gemm<0, true ><<<g, dim3(128), 0, st>>>(A, w, b, res,     C, M, Kd, N);
    else          k_gemm<0, false><<<g, dim3(128), 0, st>>>(A, w, b, nullptr, C, M, Kd, N);
}

extern "C" void kernel_launch(void* const* d_in, const int* in_sizes, int n_in,
                              void* d_out, int out_size, void* d_ws, size_t ws_size,
                              hipStream_t stream)
{
    (void)in_sizes; (void)n_in; (void)out_size; (void)ws_size;

    const float* H      = (const float*)d_in[0];
    const float* assign = (const float*)d_in[1];
    const float* E      = (const float*)d_in[2];

    // Params in jax pytree flatten order (dict keys alphabetical):
    // top level: cls, inter, intra, read ; {'w','b'} -> b,w ; {'g','b'} -> b,g
    int pi = 3;
    auto NF = [&]() -> const float* { return (const float*)d_in[pi++]; };

    // cls: cls0, e2d, ffn(fc1,fc2), ln, wk, wq, wv
    const float* p_cls0 = NF();
    LinP c_e2d; c_e2d.b = NF(); c_e2d.w = NF();
    LinP c_f1;  c_f1.b  = NF(); c_f1.w  = NF();
    LinP c_f2;  c_f2.b  = NF(); c_f2.w  = NF();
    LNPm c_ln;  c_ln.b  = NF(); c_ln.g  = NF();
    LinP c_wk;  c_wk.b  = NF(); c_wk.w  = NF();
    LinP c_wq;  c_wq.b  = NF(); c_wq.w  = NF();
    LinP c_wv;  c_wv.b  = NF(); c_wv.w  = NF();

    // inter[0]: ffn, ln1, ln2, proj, wk, wq, wv
    LinP i_f1;  i_f1.b  = NF(); i_f1.w  = NF();
    LinP i_f2;  i_f2.b  = NF(); i_f2.w  = NF();
    LNPm i_ln1; i_ln1.b = NF(); i_ln1.g = NF();
    LNPm i_ln2; i_ln2.b = NF(); i_ln2.g = NF();
    LinP i_pr;  i_pr.b  = NF(); i_pr.w  = NF();
    LinP i_wk;  i_wk.b  = NF(); i_wk.w  = NF();
    LinP i_wq;  i_wq.b  = NF(); i_wq.w  = NF();
    LinP i_wv;  i_wv.b  = NF(); i_wv.w  = NF();

    // intra[0..1]: ffn, film1(fc1,fc2), film2(fc1,fc2), ln1, ln2, proj, wk, wq, wv
    struct IntraP { LinP f1, f2, m1a, m1b, m2a, m2b; LNPm ln1, ln2; LinP pr, wk, wq, wv; } ip[2];
    for (int l = 0; l < 2; ++l) {
        ip[l].f1.b  = NF(); ip[l].f1.w  = NF();
        ip[l].f2.b  = NF(); ip[l].f2.w  = NF();
        ip[l].m1a.b = NF(); ip[l].m1a.w = NF();
        ip[l].m1b.b = NF(); ip[l].m1b.w = NF();
        ip[l].m2a.b = NF(); ip[l].m2a.w = NF();
        ip[l].m2b.b = NF(); ip[l].m2b.w = NF();
        ip[l].ln1.b = NF(); ip[l].ln1.g = NF();
        ip[l].ln2.b = NF(); ip[l].ln2.g = NF();
        ip[l].pr.b  = NF(); ip[l].pr.w  = NF();
        ip[l].wk.b  = NF(); ip[l].wk.w  = NF();
        ip[l].wq.b  = NF(); ip[l].wq.w  = NF();
        ip[l].wv.b  = NF(); ip[l].wv.w  = NF();
    }

    // read: g, ln, wk, wq, wv
    const float* r_g = NF();
    LNPm r_ln; r_ln.b = NF(); r_ln.g = NF();
    LinP r_wk; r_wk.b = NF(); r_wk.w = NF();
    LinP r_wq; r_wq.b = NF(); r_wq.w = NF();
    LinP r_wv; r_wv.b = NF(); r_wv.w = NF();

    // Workspace carve-up
    float* W = (float*)d_ws;
    size_t off = 0;
    auto alloc = [&](size_t n) { float* p = W + off; off += n; return p; };
    const int BV = cB * cV, BK = cB * cK;
    float* x     = alloc((size_t)BV * cD);
    float* yb    = alloc((size_t)BV * cD);
    float* conn  = alloc((size_t)cB * cV * cV);
    float* enode = alloc((size_t)BV * cDE);
    float* big   = alloc((size_t)BV * cDFF);         // 64 MB reusable region
    float* clsS  = alloc((size_t)cK * cD);
    float* clsB  = alloc((size_t)BK * cD);
    float* mbuf  = alloc((size_t)BK * cD);
    float* sA    = alloc((size_t)BK * cD);
    float* sB    = alloc((size_t)BK * cD);
    float* sC    = alloc((size_t)BK * cD);
    float* sD    = alloc((size_t)BK * cD);
    float* t2048 = alloc((size_t)BK * cDFF);
    float* cmask = alloc((size_t)cB * cK * cV);
    float* gq    = alloc((size_t)cB * cD);
    float* lng   = alloc((size_t)cB * cD);
    float* qr    = alloc((size_t)cB * cD);

    float* bigA = big;                         // B*V*1024 slot
    float* bigB = big + (size_t)BV * 1024;     // B*V*1024 slot
    float* qg   = big;
    float* kg   = big + (size_t)BV * cD;
    float* vg   = big + (size_t)BV * cD * 2;
    float* ao   = big + (size_t)BV * cD * 3;

    const float sc64  = 0.125f;                 // 1/sqrt(64)
    const float sc128 = 0.0883883476f;          // 1/sqrt(128)

    // x = H
    hipMemcpyAsync(x, H, (size_t)BV * cD * sizeof(float), hipMemcpyDeviceToDevice, stream);

    // conn log-mask and e_node
    {
        size_t tot = (size_t)cB * cV * cV;
        k_conn<<<(tot + 255) / 256, 256, 0, stream>>>(assign, conn, cB, cV, cK);
    }
    gemm(stream, assign, E, nullptr, nullptr, enode, BV, cK, cDE, 0);

    // ---- intra layers ----
    for (int l = 0; l < 2; ++l) {
        IntraP& P = ip[l];
        // FiLM1
        gemm(stream, enode, P.m1a.w, P.m1a.b, nullptr, bigA, BV, cDE, 2 * cD, 1);
        gemm(stream, bigA,  P.m1b.w, P.m1b.b, nullptr, bigB, BV, 2 * cD, 2 * cD, 0);
        k_ln<<<BV, 128, 0, stream>>>(x, P.ln1.g, P.ln1.b, bigB, yb, BV, cD);
        // QKV
        gemm(stream, yb, P.wq.w, P.wq.b, nullptr, qg, BV, cD, cD, 0);
        gemm(stream, yb, P.wk.w, P.wk.b, nullptr, kg, BV, cD, cD, 0);
        gemm(stream, yb, P.wv.w, P.wv.b, nullptr, vg, BV, cD, cD, 0);
        k_attn<64, true><<<dim3(cV / 64, 8, cB), 128, 0, stream>>>(
            qg, kg, vg, conn, ao, cB, 8, cV, cV, sc64, 1e-8f);
        gemm(stream, ao, P.pr.w, P.pr.b, x, x, BV, cD, cD, 0);
        // FiLM2 + FFN
        gemm(stream, enode, P.m2a.w, P.m2a.b, nullptr, bigA, BV, cDE, 2 * cD, 1);
        gemm(stream, bigA,  P.m2b.w, P.m2b.b, nullptr, bigB, BV, 2 * cD, 2 * cD, 0);
        k_ln<<<BV, 128, 0, stream>>>(x, P.ln2.g, P.ln2.b, bigB, yb, BV, cD);
        gemm(stream, yb,  P.f1.w, P.f1.b, nullptr, big, BV, cD, cDFF, 1);
        gemm(stream, big, P.f2.w, P.f2.b, x, x, BV, cDFF, cD, 0);
    }

    // ---- cls pooling ----
    gemm(stream, E, c_e2d.w, c_e2d.b, p_cls0, clsS, cK, cDE, cD, 0);
    {
        size_t tot = (size_t)BK * cD;
        k_bcast<<<(tot + 255) / 256, 256, 0, stream>>>(clsB, clsS, tot, cK * cD);
    }
    k_ln<<<BK, 128, 0, stream>>>(clsB, c_ln.g, c_ln.b, nullptr, sA, BK, cD);
    gemm(stream, sA, c_wq.w, c_wq.b, nullptr, sB, BK, cD, cD, 0);
    gemm(stream, x,  c_wk.w, c_wk.b, nullptr, qg, BV, cD, cD, 0);   // reuse big: K
    gemm(stream, x,  c_wv.w, c_wv.b, nullptr, kg, BV, cD, cD, 0);   // reuse big: V
    {
        size_t tot = (size_t)cB * cK * cV;
        k_clsmask<<<(tot + 255) / 256, 256, 0, stream>>>(assign, cmask, cB, cK, cV);
    }
    k_attn<64, true><<<dim3(1, 8, cB), 128, 0, stream>>>(
        sB, qg, kg, cmask, mbuf, cB, 8, cK, cV, sc64, 1e-8f);
    k_empty_fix<<<BK, 128, 0, stream>>>(assign, clsB, mbuf, cB, cK, cV, cD);
    k_ln<<<BK, 128, 0, stream>>>(mbuf, c_ln.g, c_ln.b, nullptr, sA, BK, cD);
    gemm(stream, sA,    c_f1.w, c_f1.b, nullptr, t2048, BK, cD, cDFF, 1);
    gemm(stream, t2048, c_f2.w, c_f2.b, mbuf, mbuf, BK, cDFF, cD, 0);

    // ---- inter layer ----
    k_ln<<<BK, 128, 0, stream>>>(mbuf, i_ln1.g, i_ln1.b, nullptr, sA, BK, cD);
    gemm(stream, sA,   i_wq.w, i_wq.b, nullptr, sB, BK, cD, cD, 0);
    gemm(stream, mbuf, i_wk.w, i_wk.b, nullptr, sC, BK, cD, cD, 0);
    gemm(stream, mbuf, i_wv.w, i_wv.b, nullptr, sD, BK, cD, cD, 0);
    k_attn<64, false><<<dim3(1, 8, cB), 128, 0, stream>>>(
        sB, sC, sD, (const float*)nullptr, sA, cB, 8, cK, cK, sc64, 0.0f);
    gemm(stream, sA, i_pr.w, i_pr.b, mbuf, mbuf, BK, cD, cD, 0);
    k_ln<<<BK, 128, 0, stream>>>(mbuf, i_ln2.g, i_ln2.b, nullptr, sA, BK, cD);
    gemm(stream, sA,    i_f1.w, i_f1.b, nullptr, t2048, BK, cD, cDFF, 1);
    gemm(stream, t2048, i_f2.w, i_f2.b, mbuf, mbuf, BK, cDFF, cD, 0);

    // ---- readout ----
    {
        size_t tot = (size_t)cB * cD;
        k_bcast<<<(tot + 255) / 256, 256, 0, stream>>>(gq, r_g, tot, cD);
    }
    k_ln<<<cB, 128, 0, stream>>>(gq, r_ln.g, r_ln.b, nullptr, lng, cB, cD);
    gemm(stream, lng,  r_wq.w, r_wq.b, nullptr, qr, cB, cD, cD, 0);
    gemm(stream, mbuf, r_wk.w, r_wk.b, nullptr, sB, BK, cD, cD, 0);
    gemm(stream, mbuf, r_wv.w, r_wv.b, nullptr, sC, BK, cD, cD, 0);
    k_attn<128, false><<<dim3(1, 4, cB), 128, 0, stream>>>(
        qr, sB, sC, (const float*)nullptr, (float*)d_out, cB, 4, 1, cK, sc128, 0.0f);
}
